// GNNCWT2D_Mk11_1sec_43989055046101
// MI455X (gfx1250) — compile-verified, matrix operand order checked by reading, not run on hardware
//
#include <hip/hip_runtime.h>
#include <hip/hip_bf16.h>

// GNN forward for MI455X (gfx1250, wave32).
// Roofline: dominated by the 389MB read of x (windowed mean) ~17us @ 23.3TB/s.
// All post-mean tensors (<32MB total) are L2-resident (192MB L2), so GEMMs load
// fragments straight from global and use v_wmma_f32_16x16x32_f16 (f16 in, f32 acc).

typedef _Float16 f16;
typedef _Float16 v16h __attribute__((ext_vector_type(16)));
typedef _Float16 v8h  __attribute__((ext_vector_type(8)));
typedef float    v8f  __attribute__((ext_vector_type(8)));

constexpr int BQ = 256, NEL = 19, NFEAT = 40, NTIME = 500, TWIN = 25, NWIN = 20;
constexpr int NNODES = BQ * NEL;          // 4864 (= 304 * 16)
constexpr int F0 = NFEAT * NWIN;          // 800
constexpr int EPER = 60, NEDGE = BQ * EPER;

enum { EPI_BN_NODE = 0, EPI_BN_FEAT = 1, EPI_RAW = 2 };

// ---------------------------------------------------------------- fragment load
// 16-bit A/B fragment per CDNA5 ISA (7.12.2): lanes 0-15 -> row (lane&15),
// K halves 0..7 & 16..23; lanes 16-31 -> K halves 8..15 & 24..31.
__device__ __forceinline__ v16h ld_frag(const f16* p) {
  v8h lo = *reinterpret_cast<const v8h*>(p);
  v8h hi = *reinterpret_cast<const v8h*>(p + 16);
  v16h r;
#pragma unroll
  for (int i = 0; i < 8; ++i) { r[i] = lo[i]; r[i + 8] = hi[i]; }
  return r;
}

// ---------------------------------------------------------------- stage 1: mean
// h0[n, f*NW + w] = mean over 25 time samples; f16 output for WMMA consumption.
__global__ __launch_bounds__(256)
void win_mean_kernel(const float* __restrict__ x, f16* __restrict__ h0) {
  int gid = blockIdx.x * 256 + threadIdx.x;
  if (gid >= NNODES * F0) return;
  int n = gid / F0, fw = gid - n * F0;
  int f = fw / NWIN, w = fw - f * NWIN;
  const float* p = x + (size_t)n * (NFEAT * NTIME) + f * NTIME + w * TWIN;
  float s = 0.f;
#pragma unroll
  for (int t = 0; t < TWIN; ++t) s += p[t];
  h0[gid] = (f16)(s * (1.0f / TWIN));
}

__global__ __launch_bounds__(256)
void f32_to_f16_kernel(const float* __restrict__ a, f16* __restrict__ b, int n) {
  int gid = blockIdx.x * 256 + threadIdx.x;
  if (gid < n) b[gid] = (f16)a[gid];
}

// ---------------------------------------------------------------- WMMA GEMM
// C = A @ W^T (+ A2 @ W2^T) with fused epilogue.  A:[M,K] f16 row-major,
// W:[N,K] f16 row-major (so B fragment = rows of W).  One wave -> 16x64 tile.
template <int MODE, bool DUAL>
__global__ __launch_bounds__(256)
void gemm_wmma_kernel(const f16* __restrict__ A, const f16* __restrict__ W,
                      const f16* __restrict__ A2, const f16* __restrict__ W2,
                      const float* __restrict__ bias, const float* __restrict__ bn,
                      int bnC, float* __restrict__ outF, f16* __restrict__ outH,
                      int M, int N, int K) {
  const int lane = threadIdx.x & 31;
  const int wave = blockIdx.x * (blockDim.x >> 5) + (threadIdx.x >> 5);
  const int tilesN = N >> 6;
  const int mt = wave / tilesN, nt = wave - mt * tilesN;
  const int m0 = mt << 4, n0 = nt << 6;
  if (m0 >= M) return;                   // grids are exact; whole-wave guard only
  const int r  = lane & 15;
  const int ks = (lane >> 4) << 3;       // K-half select: 0 or 8

  v8f acc[4];
#pragma unroll
  for (int t = 0; t < 4; ++t) {
#pragma unroll
    for (int i = 0; i < 8; ++i) acc[t][i] = 0.0f;
  }

  {
    const f16* ap = A + (size_t)(m0 + r) * K + ks;
    const f16* bp = W + (size_t)(n0 + r) * K + ks;
    for (int k0 = 0; k0 < K; k0 += 32) {
      v16h a = ld_frag(ap + k0);
#pragma unroll
      for (int t = 0; t < 4; ++t) {
        v16h b = ld_frag(bp + (size_t)(t * 16) * K + k0);
        acc[t] = __builtin_amdgcn_wmma_f32_16x16x32_f16(
            false, a, false, b, (short)0, acc[t], false, false);
      }
    }
  }
  if (DUAL) {
    const f16* ap = A2 + (size_t)(m0 + r) * K + ks;
    const f16* bp = W2 + (size_t)(n0 + r) * K + ks;
    for (int k0 = 0; k0 < K; k0 += 32) {
      v16h a = ld_frag(ap + k0);
#pragma unroll
      for (int t = 0; t < 4; ++t) {
        v16h b = ld_frag(bp + (size_t)(t * 16) * K + k0);
        acc[t] = __builtin_amdgcn_wmma_f32_16x16x32_f16(
            false, a, false, b, (short)0, acc[t], false, false);
      }
    }
  }

  const int mofs = (lane >> 4) << 3;     // C layout: lanes>=16 hold M rows 8..15
#pragma unroll
  for (int t = 0; t < 4; ++t) {
#pragma unroll
    for (int v = 0; v < 8; ++v) {
      int m = m0 + mofs + v;
      int n = n0 + t * 16 + r;
      float val = acc[t][v];
      if (MODE != EPI_RAW) {
        val += bias[n];
        val = fmaxf(val, 0.0f);          // ReLU before eval-mode BN
        int c = (MODE == EPI_BN_NODE) ? (m % NEL) : n;
        float g  = bn[c],           be = bn[bnC + c];
        float mu = bn[2 * bnC + c], va = bn[3 * bnC + c];
        val = (val - mu) * (g * rsqrtf(va + 1e-5f)) + be;
      }
      size_t o = (size_t)m * N + n;
      if (outF) outF[o] = val;
      if (outH) outH[o] = (f16)val;
    }
  }
}

// ---------------------------------------------------------------- GraphConv agg
// agg[n,f] = sum_{edges j with dst==n} softplus(edge_w[j]) * h3[src_j, f]
__global__ __launch_bounds__(256)
void graphconv_agg_kernel(const int* __restrict__ srcI, const int* __restrict__ dstI,
                          const float* __restrict__ ew, const f16* __restrict__ h3,
                          f16* __restrict__ aggH) {
  __shared__ float sw[EPER];
  int t = threadIdx.x;
  if (t < EPER) {
    float xw = ew[t];
    sw[t] = (xw > 20.f) ? xw : log1pf(__expf(xw));
  }
  __syncthreads();
  int gid = blockIdx.x * 256 + t;
  int n = gid >> 7, f = gid & 127;
  int b = n / NEL;
  const int* sp = srcI + b * EPER;
  const int* dp = dstI + b * EPER;
  float acc = 0.f;
  for (int j = 0; j < EPER; ++j)
    if (dp[j] == n) acc += sw[j] * (float)h3[(size_t)sp[j] * 128 + f];
  aggH[gid] = (f16)acc;
}

// ---------------------------------------------------------------- GAT attention
__global__ __launch_bounds__(256)
void gat_attn_prep_kernel(const float* __restrict__ xh, const float* __restrict__ att_s,
                          const float* __restrict__ att_d, float* __restrict__ as,
                          float* __restrict__ ad) {
  int gid = blockIdx.x * 256 + threadIdx.x;
  int n = gid >> 2, hd = gid & 3;
  const float* xp = xh + (size_t)n * 256 + hd * 64;
  const float* spv = att_s + hd * 64;
  const float* dpv = att_d + hd * 64;
  float s = 0.f, d = 0.f;
#pragma unroll 8
  for (int c = 0; c < 64; ++c) { float v = xp[c]; s += v * spv[c]; d += v * dpv[c]; }
  as[gid] = s;
  ad[gid] = d;
}

__device__ __forceinline__ float lrelu(float x) { return x > 0.f ? x : 0.2f * x; }

// Per (dst node, head, channel): segment softmax over {incoming edges} U {self-loop}.
__global__ __launch_bounds__(256)
void gat_aggregate_kernel(const int* __restrict__ srcI, const int* __restrict__ dstI,
                          const float* __restrict__ as, const float* __restrict__ ad,
                          const float* __restrict__ xh, float* __restrict__ go) {
  int gid = blockIdx.x * 256 + threadIdx.x;
  int n = gid >> 8, hc = gid & 255;
  int hd = hc >> 6, c = hc & 63;
  int b = n / NEL;
  const int* sp = srcI + b * EPER;
  const int* dp = dstI + b * EPER;
  float adv = ad[n * 4 + hd];
  float e_self = lrelu(as[n * 4 + hd] + adv);
  float mx = e_self;
  for (int j = 0; j < EPER; ++j)
    if (dp[j] == n) mx = fmaxf(mx, lrelu(as[sp[j] * 4 + hd] + adv));
  float wsum = __expf(e_self - mx);
  float acc = wsum * xh[(size_t)n * 256 + hd * 64 + c];
  for (int j = 0; j < EPER; ++j)
    if (dp[j] == n) {
      float w = __expf(lrelu(as[sp[j] * 4 + hd] + adv) - mx);
      wsum += w;
      acc  += w * xh[(size_t)sp[j] * 256 + hd * 64 + c];
    }
  go[gid] = acc / (wsum + 1e-16f);
}

// head-mean + bias + ReLU + BN(feature) + global max pool over 19 nodes/graph
__global__ __launch_bounds__(256)
void headmean_bn_pool_kernel(const float* __restrict__ go, const float* __restrict__ gbias,
                             const float* __restrict__ bn, float* __restrict__ pooled) {
  int gid = blockIdx.x * 256 + threadIdx.x;
  int b = gid >> 6, c = gid & 63;
  float inv = bn[c] * rsqrtf(bn[192 + c] + 1e-5f);
  float be = bn[64 + c], mu = bn[128 + c];
  float bias = gbias[c];
  float mx = -3.402823466e+38f;
  for (int e = 0; e < NEL; ++e) {
    size_t base = (size_t)(b * NEL + e) * 256 + c;
    float s = 0.25f * (go[base] + go[base + 64] + go[base + 128] + go[base + 192]) + bias;
    s = fmaxf(s, 0.f);
    s = (s - mu) * inv + be;
    mx = fmaxf(mx, s);
  }
  pooled[gid] = mx;
}

__global__ __launch_bounds__(256)
void final_linear_kernel(const float* __restrict__ pooled, const float* __restrict__ w,
                         const float* __restrict__ bias, float* __restrict__ out) {
  int gid = blockIdx.x * 256 + threadIdx.x;
  int b = gid >> 6, o = gid & 63;
  const float* pp = pooled + b * 64;
  const float* wp = w + o * 64;
  float s = bias[o];
#pragma unroll 8
  for (int k = 0; k < 64; ++k) s += pp[k] * wp[k];
  out[gid] = fmaxf(s, 0.f);
}

// ---------------------------------------------------------------- launch
extern "C" void kernel_launch(void* const* d_in, const int* in_sizes, int n_in,
                              void* d_out, int out_size, void* d_ws, size_t ws_size,
                              hipStream_t stream) {
  (void)in_sizes; (void)n_in; (void)out_size; (void)ws_size;
  const float* x        = (const float*)d_in[0];
  const int*   ei       = (const int*)d_in[1];
  const int*   srcI     = ei;
  const int*   dstI     = ei + NEDGE;
  const float* lin2_w   = (const float*)d_in[3];
  const float* lin2_b   = (const float*)d_in[4];
  const float* lin3_w   = (const float*)d_in[5];
  const float* lin3_b   = (const float*)d_in[6];
  const float* lin4_w   = (const float*)d_in[7];
  const float* lin4_b   = (const float*)d_in[8];
  const float* bn2      = (const float*)d_in[9];
  const float* bn3      = (const float*)d_in[10];
  const float* bn4      = (const float*)d_in[11];
  const float* edge_w   = (const float*)d_in[12];
  const float* g1_rel_w = (const float*)d_in[13];
  const float* g1_rel_b = (const float*)d_in[14];
  const float* g1_root_w= (const float*)d_in[15];
  const float* bn_g1    = (const float*)d_in[16];
  const float* gat_w    = (const float*)d_in[17];
  const float* att_src  = (const float*)d_in[18];
  const float* att_dst  = (const float*)d_in[19];
  const float* gat_bias = (const float*)d_in[20];
  const float* bn_g2    = (const float*)d_in[21];
  const float* lfw      = (const float*)d_in[22];
  const float* lfb      = (const float*)d_in[23];
  float* out = (float*)d_out;

  char* ws = (char*)d_ws;
  auto alloc = [&](size_t bytes) -> void* {
    void* p = ws;
    ws += (bytes + 255) & ~(size_t)255;
    return p;
  };
  f16*   h0   = (f16*)alloc((size_t)NNODES * F0 * 2);
  f16*   w2h  = (f16*)alloc((size_t)512 * 800 * 2);
  f16*   w3h  = (f16*)alloc((size_t)256 * 512 * 2);
  f16*   w4h  = (f16*)alloc((size_t)128 * 256 * 2);
  f16*   wr1h = (f16*)alloc((size_t)64 * 128 * 2);
  f16*   wr2h = (f16*)alloc((size_t)64 * 128 * 2);
  f16*   wgh  = (f16*)alloc((size_t)256 * 64 * 2);
  f16*   h1   = (f16*)alloc((size_t)NNODES * 512 * 2);
  f16*   h2   = (f16*)alloc((size_t)NNODES * 256 * 2);
  f16*   h3   = (f16*)alloc((size_t)NNODES * 128 * 2);
  f16*   aggH = (f16*)alloc((size_t)NNODES * 128 * 2);
  f16*   h4   = (f16*)alloc((size_t)NNODES * 64 * 2);
  float* xh   = (float*)alloc((size_t)NNODES * 256 * 4);
  float* asb  = (float*)alloc((size_t)NNODES * 4 * 4);
  float* adb  = (float*)alloc((size_t)NNODES * 4 * 4);
  float* go   = (float*)alloc((size_t)NNODES * 256 * 4);
  float* pooled = (float*)alloc((size_t)BQ * 64 * 4);

  // stage 1: bandwidth-bound windowed mean (389MB -> 7.8MB f16)
  win_mean_kernel<<<(NNODES * F0) / 256, 256, 0, stream>>>(x, h0);

  // weight conversions (tiny, L2-resident afterwards)
  f32_to_f16_kernel<<<(512 * 800 + 255) / 256, 256, 0, stream>>>(lin2_w, w2h, 512 * 800);
  f32_to_f16_kernel<<<(256 * 512 + 255) / 256, 256, 0, stream>>>(lin3_w, w3h, 256 * 512);
  f32_to_f16_kernel<<<(128 * 256 + 255) / 256, 256, 0, stream>>>(lin4_w, w4h, 128 * 256);
  f32_to_f16_kernel<<<(64 * 128 + 255) / 256, 256, 0, stream>>>(g1_rel_w, wr1h, 64 * 128);
  f32_to_f16_kernel<<<(64 * 128 + 255) / 256, 256, 0, stream>>>(g1_root_w, wr2h, 64 * 128);
  f32_to_f16_kernel<<<(256 * 64 + 255) / 256, 256, 0, stream>>>(gat_w, wgh, 256 * 64);

  const int MT = NNODES / 16;  // 304 M-tiles
  // lin2: 4864x800 @ 800x512, ReLU + BN over electrode
  gemm_wmma_kernel<EPI_BN_NODE, false><<<MT * (512 / 64) / 8, 256, 0, stream>>>(
      h0, w2h, nullptr, nullptr, lin2_b, bn2, NEL, nullptr, h1, NNODES, 512, 800);
  // lin3
  gemm_wmma_kernel<EPI_BN_NODE, false><<<MT * (256 / 64) / 8, 256, 0, stream>>>(
      h1, w3h, nullptr, nullptr, lin3_b, bn3, NEL, nullptr, h2, NNODES, 256, 512);
  // lin4
  gemm_wmma_kernel<EPI_BN_NODE, false><<<MT * (128 / 64) / 8, 256, 0, stream>>>(
      h2, w4h, nullptr, nullptr, lin4_b, bn4, NEL, nullptr, h3, NNODES, 128, 256);

  // GraphConv aggregation (fixed 60-edge topology -> per-dst gather)
  graphconv_agg_kernel<<<(NNODES * 128) / 256, 256, 0, stream>>>(srcI, dstI, edge_w, h3, aggH);

  // fused GraphConv GEMM: relu(agg@Wrel^T + h3@Wroot^T + b), BN over 64 features
  gemm_wmma_kernel<EPI_BN_FEAT, true><<<MT * (64 / 64) / 8, 256, 0, stream>>>(
      aggH, wr1h, h3, wr2h, g1_rel_b, bn_g1, 64, nullptr, h4, NNODES, 64, 128);

  // GAT projection: xh = h4 @ gat_w^T  (raw f32 out for attention math)
  gemm_wmma_kernel<EPI_RAW, false><<<MT * (256 / 64) / 8, 256, 0, stream>>>(
      h4, wgh, nullptr, nullptr, nullptr, nullptr, 0, xh, nullptr, NNODES, 256, 64);

  gat_attn_prep_kernel<<<(NNODES * 4) / 256, 256, 0, stream>>>(xh, att_src, att_dst, asb, adb);
  gat_aggregate_kernel<<<(NNODES * 256) / 256, 256, 0, stream>>>(srcI, dstI, asb, adb, xh, go);
  headmean_bn_pool_kernel<<<(BQ * 64) / 256, 256, 0, stream>>>(go, gat_bias, bn_g2, pooled);
  final_linear_kernel<<<(BQ * 64) / 256, 256, 0, stream>>>(pooled, lfw, lfb, out);
}
